// Decoder_24713241821884
// MI455X (gfx1250) — matrix operand order repaired and checked by reading, not compiled, not run on hardware
//
#include <hip/hip_runtime.h>
#include <hip/hip_bf16.h>
#include <math.h>

typedef __attribute__((ext_vector_type(16))) __bf16 v16bf;
typedef __attribute__((ext_vector_type(8)))  float  v8f;

union BF16x16 { v16bf v; unsigned short u[16]; uint4 q[2]; };

__device__ __forceinline__ unsigned short f2bf(float f) {
  union { float f; unsigned u; } x; x.f = f;
  unsigned r = x.u + 0x7FFFu + ((x.u >> 16) & 1u);   // round-to-nearest-even
  return (unsigned short)(r >> 16);
}
__device__ __forceinline__ void cvt4(unsigned short* d, float4 f) {
  d[0] = f2bf(f.x); d[1] = f2bf(f.y); d[2] = f2bf(f.z); d[3] = f2bf(f.w);
}

// ---------------------------------------------------------------------------
// Prep: repack W_e (columns H..2H of attn_W) into bf16 B-fragment order.
// Fragment flat index: (((e*32+nt)*16+kt)*32+lane)*16 + i
// B layout (32x16 bf16): lane ln=lane&15 -> column n=nt*16+ln, hi=lane>>4,
//                        element i -> K = kt*32 + i + 16*hi.
// ---------------------------------------------------------------------------
__global__ void k_prep_w(const float* __restrict__ ws, const float* __restrict__ wc,
                         const float* __restrict__ wa, unsigned short* __restrict__ prepW) {
  int flat = blockIdx.x * blockDim.x + threadIdx.x;
  if (flat >= 3 * 512 * 512) return;
  int i    =  flat        & 15;
  int lane = (flat >> 4)  & 31;
  int kt   = (flat >> 9)  & 15;
  int nt   = (flat >> 13) & 31;
  int e    =  flat >> 18;
  int n = nt * 16 + (lane & 15);
  int k = kt * 32 + i + 16 * (lane >> 4);
  const float* w = (e == 0) ? ws : (e == 1) ? wc : wa;
  prepW[flat] = f2bf(w[n * 1024 + 512 + k]);
}

__global__ void k_embed(const int* __restrict__ inputs, const float* __restrict__ emb,
                        float* __restrict__ embedded) {
  int b = blockIdx.x;
  long row = inputs[b];
  for (int c = threadIdx.x; c < 512; c += blockDim.x)
    embedded[b * 512 + c] = emb[row * 512 + c];
}

// hpart[e][b][n] = attn_b[n] + sum_k W[n,k]*h_prev[b,k]   (first H columns)
__global__ void k_hpart(const float* __restrict__ h_prev,
                        const float* __restrict__ Ws, const float* __restrict__ bs,
                        const float* __restrict__ Wc, const float* __restrict__ bc,
                        const float* __restrict__ Wa, const float* __restrict__ ba,
                        float* __restrict__ hpart) {
  __shared__ float hs[512];
  int e = blockIdx.x >> 6, b = blockIdx.x & 63;
  const float* W  = (e == 0) ? Ws : (e == 1) ? Wc : Wa;
  const float* bb = (e == 0) ? bs : (e == 1) ? bc : ba;
  for (int k = threadIdx.x; k < 512; k += blockDim.x) hs[k] = h_prev[b * 512 + k];
  __syncthreads();
  for (int n = threadIdx.x; n < 512; n += blockDim.x) {
    float acc = bb[n];
    const float* wr = W + (long)n * 1024;
#pragma unroll 4
    for (int k = 0; k < 512; ++k) acc += wr[k] * hs[k];
    hpart[((e * 64 + b) << 9) + n] = acc;
  }
}

// ---------------------------------------------------------------------------
// Attention scores: score[b,t] = sum_n v[n]*relu( (enc_row . We_col_n) + hpart[b,n] )
// Rows r = t*64+b are contiguous 512-f32 vectors. One wave = one 16-row M-tile,
// A fragments for all K=512 kept in VGPRs (bf16), 32 N-tiles x 16 WMMAs each.
// ---------------------------------------------------------------------------
__global__ __launch_bounds__(256)
void k_attn_scores(const float* __restrict__ enc_s, const float* __restrict__ enc_c,
                   const float* __restrict__ enc_a,
                   const unsigned short* __restrict__ prepW,
                   const float* __restrict__ hpart,
                   const float* __restrict__ vs, const float* __restrict__ vc,
                   const float* __restrict__ va,
                   float* __restrict__ scores) {
  const int e   = blockIdx.y;
  const float* enc = (e == 0) ? enc_s : (e == 1) ? enc_c : enc_a;
  const float* v   = (e == 0) ? vs   : (e == 1) ? vc   : va;
  const int wave = threadIdx.x >> 5;
  const int lane = threadIdx.x & 31;
  const int ln = lane & 15, hi = lane >> 4;
  const int r0 = (blockIdx.x * 8 + wave) * 16;          // rows r0..r0+15, r = t*64+b

  // --- load all A fragments (16 k-steps), convert f32 -> bf16, contiguous k layout (K=i+16hi) ---
  BF16x16 aF[16];
  {
    const long rowbase = (long)(r0 + ln) * 512;
#pragma unroll
    for (int kt = 0; kt < 16; ++kt) {
      const float* p0 = enc + rowbase + kt * 32 + 16 * hi;     // elems 0..7 : K=i+16hi
      const float* p1 = p0 + 8;                                // elems 8..15: K=i+16hi
      float4 f0 = ((const float4*)p0)[0];
      float4 f1 = ((const float4*)p0)[1];
      float4 f2 = ((const float4*)p1)[0];
      float4 f3 = ((const float4*)p1)[1];
      cvt4(&aF[kt].u[0],  f0); cvt4(&aF[kt].u[4],  f1);
      cvt4(&aF[kt].u[8],  f2); cvt4(&aF[kt].u[12], f3);
    }
  }

  // hpart row pointers for the 8 (m) rows this lane-half covers
  const float* hp[8];
#pragma unroll
  for (int j = 0; j < 8; ++j)
    hp[j] = hpart + ((e * 64 + ((r0 + j + 8 * hi) & 63)) << 9);

  float part[8] = {0.f, 0.f, 0.f, 0.f, 0.f, 0.f, 0.f, 0.f};

  for (int nt = 0; nt < 32; ++nt) {
    v8f acc = {};
#pragma unroll
    for (int kt = 0; kt < 16; ++kt) {
      BF16x16 bF;
      const uint4* q = ((const uint4*)prepW) + ((long)(((e * 32 + nt) * 16 + kt) * 32 + lane)) * 2;
      bF.q[0] = q[0]; bF.q[1] = q[1];
      acc = __builtin_amdgcn_wmma_f32_16x16x32_bf16(false, aF[kt].v, false, bF.v,
                                                    (short)0, acc, false, false);
    }
    const int n = nt * 16 + ln;
    const float vv = v[n];
#pragma unroll
    for (int j = 0; j < 8; ++j) {
      float en = acc[j] + hp[j][n];
      part[j] += (en > 0.f ? en : 0.f) * vv;
    }
  }

#pragma unroll
  for (int j = 0; j < 8; ++j) {
    float s = part[j];
    s += __shfl_xor(s, 1); s += __shfl_xor(s, 2);
    s += __shfl_xor(s, 4); s += __shfl_xor(s, 8);      // reduce within 16-lane half
    if (ln == 0) {
      int r = r0 + j + 8 * hi;
      int t = r >> 6, b = r & 63;
      scores[((e * 64 + b) << 9) + t] = s;
    }
  }
}

__global__ void k_softmax_t(const float* __restrict__ scores, float* __restrict__ wout) {
  __shared__ float red[256];
  int e = blockIdx.y, b = blockIdx.x;
  const float* s = scores + ((e * 64 + b) << 9);
  float m = -1e30f;
  for (int t = threadIdx.x; t < 512; t += 256) m = fmaxf(m, s[t]);
  red[threadIdx.x] = m; __syncthreads();
  for (int o = 128; o; o >>= 1) { if (threadIdx.x < o) red[threadIdx.x] = fmaxf(red[threadIdx.x], red[threadIdx.x + o]); __syncthreads(); }
  m = red[0]; __syncthreads();
  float ex[2]; float sum = 0.f;
  { int i = 0; for (int t = threadIdx.x; t < 512; t += 256, ++i) { ex[i] = __expf(s[t] - m); sum += ex[i]; } }
  red[threadIdx.x] = sum; __syncthreads();
  for (int o = 128; o; o >>= 1) { if (threadIdx.x < o) red[threadIdx.x] += red[threadIdx.x + o]; __syncthreads(); }
  float inv = 1.f / red[0];
  { int i = 0; for (int t = threadIdx.x; t < 512; t += 256, ++i) wout[((e * 64 + b) << 9) + t] = ex[i] * inv; }
}

__global__ void k_zero(float* __restrict__ p, int n) {
  int i = blockIdx.x * blockDim.x + threadIdx.x;
  if (i < n) p[i] = 0.f;
}

// ctx[b,h] += coef_e * sum_t w[e,b,t]*enc_e[t,b,h] ; t split into 8 chunks
__global__ void k_context(const float* __restrict__ enc_s, const float* __restrict__ enc_c,
                          const float* __restrict__ enc_a,
                          const float* __restrict__ w, float* __restrict__ ctx) {
  int b = blockIdx.x, chunk = blockIdx.y, e = blockIdx.z;
  const float* enc = (e == 0) ? enc_s : (e == 1) ? enc_c : enc_a;
  float coef = (e == 2) ? 1.0f : 0.5f;
  int h0 = threadIdx.x, h1 = threadIdx.x + 256;
  float a0 = 0.f, a1 = 0.f;
  int t0 = chunk * 64;
  for (int t = t0; t < t0 + 64; ++t) {
    float wv = w[((e * 64 + b) << 9) + t];
    const float* row = enc + ((long)t * 64 + b) * 512;
    a0 += wv * row[h0]; a1 += wv * row[h1];
  }
  atomicAdd(&ctx[b * 512 + h0], coef * a0);
  atomicAdd(&ctx[b * 512 + h1], coef * a1);
}

__global__ void k_pgen(const float* __restrict__ ctx, const float* __restrict__ h_prev,
                       const float* __restrict__ embedded,
                       const float* __restrict__ pw, const float* __restrict__ pb,
                       float* __restrict__ pgen_out) {
  __shared__ float red[256];
  int b = blockIdx.x;
  float acc = 0.f;
  for (int i = threadIdx.x; i < 1536; i += 256) {
    float x = (i < 512) ? ctx[b * 512 + i]
            : (i < 1024) ? h_prev[b * 512 + (i - 512)]
                         : embedded[b * 512 + (i - 1024)];
    acc += x * pw[i];
  }
  red[threadIdx.x] = acc; __syncthreads();
  for (int o = 128; o; o >>= 1) { if (threadIdx.x < o) red[threadIdx.x] += red[threadIdx.x + o]; __syncthreads(); }
  if (threadIdx.x == 0) pgen_out[b] = 1.f / (1.f + __expf(-(red[0] + pb[0])));
}

__global__ void k_gru(const float* __restrict__ embedded, const float* __restrict__ ctx,
                      const float* __restrict__ h_prev,
                      const float* __restrict__ wih, const float* __restrict__ whh,
                      const float* __restrict__ bih, const float* __restrict__ bhh,
                      float* __restrict__ hidden_out, unsigned short* __restrict__ xb) {
  __shared__ float xs[1024]; __shared__ float hs[512];
  int b = blockIdx.x, hb = blockIdx.y;                 // grid (64,4), block 128
  for (int i = threadIdx.x; i < 512; i += 128) {
    xs[i] = embedded[b * 512 + i]; xs[512 + i] = ctx[b * 512 + i];
    hs[i] = h_prev[b * 512 + i];
  }
  __syncthreads();
  int h = hb * 128 + threadIdx.x;
  float ir = bih[h], iz = bih[512 + h], in_ = bih[1024 + h];
  const float* wr = wih + (long)h * 1024;
  const float* wz = wih + (long)(512 + h) * 1024;
  const float* wn = wih + (long)(1024 + h) * 1024;
#pragma unroll 4
  for (int k = 0; k < 1024; ++k) { float x = xs[k]; ir += wr[k] * x; iz += wz[k] * x; in_ += wn[k] * x; }
  float hr = bhh[h], hz = bhh[512 + h], hn = bhh[1024 + h];
  const float* ur = whh + (long)h * 512;
  const float* uz = whh + (long)(512 + h) * 512;
  const float* un = whh + (long)(1024 + h) * 512;
#pragma unroll 4
  for (int k = 0; k < 512; ++k) { float x = hs[k]; hr += ur[k] * x; hz += uz[k] * x; hn += un[k] * x; }
  float r = 1.f / (1.f + __expf(-(ir + hr)));
  float z = 1.f / (1.f + __expf(-(iz + hz)));
  float n = tanhf(in_ + r * hn);
  float hnew = (1.f - z) * n + z * hs[h];
  hidden_out[b * 512 + h] = hnew;
  xb[b * 1024 + h]       = f2bf(hnew);
  xb[b * 1024 + 512 + h] = f2bf(ctx[b * 512 + h]);
}

// ---------------------------------------------------------------------------
// Output projection: logits[64,50000] = [h_new|ctx](bf16) @ out_W^T + out_b
// One wave per 16-column vocab tile, 4 M-tiles, K=1024 (32 WMMA k-steps).
// out_W streamed once, f32->bf16 in registers.
// ---------------------------------------------------------------------------
__global__ __launch_bounds__(256)
void k_outproj(const unsigned short* __restrict__ xb, const float* __restrict__ outW,
               const float* __restrict__ outb, float* __restrict__ logits) {
  const int wave = threadIdx.x >> 5, lane = threadIdx.x & 31;
  const int ln = lane & 15, hi = lane >> 4;
  const int nt = blockIdx.x * 8 + wave;
  if (nt >= 3125) return;
  const int n0 = nt * 16;
  v8f acc[4] = {};
  const float* wrow = outW + (long)(n0 + ln) * 1024;   // this lane's column of B
#pragma unroll 1
  for (int kt = 0; kt < 32; ++kt) {
    BF16x16 bF;
    const float4* wp = (const float4*)(wrow + kt * 32 + 16 * hi);
    if (kt + 1 < 32) __builtin_prefetch(wrow + (kt + 1) * 32 + 16 * hi, 0, 0);
    float4 f0 = wp[0], f1 = wp[1], f2 = wp[2], f3 = wp[3];
    cvt4(&bF.u[0], f0); cvt4(&bF.u[4], f1); cvt4(&bF.u[8], f2); cvt4(&bF.u[12], f3);
#pragma unroll
    for (int mt = 0; mt < 4; ++mt) {
      BF16x16 aF;
      const unsigned short* xr = xb + (long)(mt * 16 + ln) * 1024 + kt * 32;
      aF.q[0] = *((const uint4*)(xr + 16 * hi));       // elems 0..7 : K=i+16hi
      aF.q[1] = *((const uint4*)(xr + 16 * hi + 8));   // elems 8..15: K=i+16hi
      acc[mt] = __builtin_amdgcn_wmma_f32_16x16x32_bf16(false, aF.v, false, bF.v,
                                                        (short)0, acc[mt], false, false);
    }
  }
  const float bb = outb[n0 + ln];
#pragma unroll
  for (int mt = 0; mt < 4; ++mt)
#pragma unroll
    for (int j = 0; j < 8; ++j) {
      int row = mt * 16 + j + 8 * hi;
      logits[(long)row * 50000 + n0 + ln] = acc[mt][j] + bb;
    }
}

__global__ void k_softmax_v(const float* __restrict__ logits, const float* __restrict__ pgen,
                            float* __restrict__ fin) {
  __shared__ float red[256];
  int b = blockIdx.x;
  const float* l = logits + (long)b * 50000;
  float m = -1e30f;
  for (int i = threadIdx.x; i < 50000; i += 256) m = fmaxf(m, l[i]);
  red[threadIdx.x] = m; __syncthreads();
  for (int o = 128; o; o >>= 1) { if (threadIdx.x < o) red[threadIdx.x] = fmaxf(red[threadIdx.x], red[threadIdx.x + o]); __syncthreads(); }
  m = red[0]; __syncthreads();
  float sum = 0.f;
  for (int i = threadIdx.x; i < 50000; i += 256) sum += __expf(l[i] - m);
  red[threadIdx.x] = sum; __syncthreads();
  for (int o = 128; o; o >>= 1) { if (threadIdx.x < o) red[threadIdx.x] += red[threadIdx.x + o]; __syncthreads(); }
  float scale = pgen[b] / red[0];
  float* f = fin + (long)b * 50050;
  for (int i = threadIdx.x; i < 50000; i += 256) f[i] = __expf(l[i] - m) * scale;
  for (int i = 50000 + threadIdx.x; i < 50050; i += 256) f[i] = 0.f;
}

__global__ void k_scatter(const int* __restrict__ ext, const float* __restrict__ srcw,
                          const float* __restrict__ pgen, float* __restrict__ fin) {
  int b = blockIdx.x, t = threadIdx.x;                 // block 512
  float coef = 1.f - pgen[b];
  int idx = ext[b * 512 + t];
  atomicAdd(&fin[(long)b * 50050 + idx], coef * srcw[b * 512 + t]);
}

__global__ void k_log(float* __restrict__ fin, long n) {
  long i = (long)blockIdx.x * blockDim.x + threadIdx.x;
  if (i < n) fin[i] = __logf(fin[i] + 1e-12f);
}

// ---------------------------------------------------------------------------
extern "C" void kernel_launch(void* const* d_in, const int* in_sizes, int n_in,
                              void* d_out, int out_size, void* d_ws, size_t ws_size,
                              hipStream_t stream) {
  const int*   inputs    = (const int*)d_in[0];
  const float* h_prev    = (const float*)d_in[1];
  const float* enc_s     = (const float*)d_in[2];
  const float* enc_c     = (const float*)d_in[3];
  const float* enc_a     = (const float*)d_in[4];
  const int*   ext       = (const int*)d_in[5];
  const float* embedding = (const float*)d_in[7];
  const float* Ws = (const float*)d_in[8];  const float* bs = (const float*)d_in[9];  const float* vs = (const float*)d_in[10];
  const float* Wc = (const float*)d_in[11]; const float* bc = (const float*)d_in[12]; const float* vc = (const float*)d_in[13];
  const float* Wa = (const float*)d_in[14]; const float* ba = (const float*)d_in[15]; const float* va = (const float*)d_in[16];
  const float* wih = (const float*)d_in[17]; const float* whh = (const float*)d_in[18];
  const float* bih = (const float*)d_in[19]; const float* bhh = (const float*)d_in[20];
  const float* outW = (const float*)d_in[21]; const float* outb = (const float*)d_in[22];
  const float* pw = (const float*)d_in[23];  const float* pb = (const float*)d_in[24];

  char* ws = (char*)d_ws;
  float*          scores   = (float*)(ws + 0);              // 3*64*512*4   = 393216
  float*          hpart    = (float*)(ws + 393216);         // 3*64*512*4   = 393216
  float*          embedded = (float*)(ws + 786432);         // 64*512*4     = 131072
  float*          ctx      = (float*)(ws + 917504);         // 64*512*4     = 131072
  unsigned short* prepW    = (unsigned short*)(ws + 1048576);// 3*512*512*2 = 1572864
  unsigned short* xb       = (unsigned short*)(ws + 2621440);// 64*1024*2   = 131072
  float*          logits   = (float*)(ws + 2752512);        // 64*50000*4   = 12800000

  float* out      = (float*)d_out;
  float* o_final  = out;                  // [64, 50050]
  float* o_hidden = out + 3203200;        // [1, 64, 512]
  float* o_w      = out + 3235968;        // src_w | code_w | ast_w, each 64*512
  float* o_pgen   = out + 3334272;        // [64, 1]

  k_prep_w<<<(3 * 512 * 512 + 255) / 256, 256, 0, stream>>>(Ws, Wc, Wa, prepW);
  k_embed<<<64, 256, 0, stream>>>(inputs, embedding, embedded);
  k_hpart<<<192, 256, 0, stream>>>(h_prev, Ws, bs, Wc, bc, Wa, ba, hpart);
  k_zero<<<128, 256, 0, stream>>>(ctx, 64 * 512);

  dim3 g1(256, 3);
  k_attn_scores<<<g1, 256, 0, stream>>>(enc_s, enc_c, enc_a, prepW, hpart, vs, vc, va, scores);

  dim3 g2(64, 3);
  k_softmax_t<<<g2, 256, 0, stream>>>(scores, o_w);

  dim3 g3(64, 8, 3);
  k_context<<<g3, 256, 0, stream>>>(enc_s, enc_c, enc_a, o_w, ctx);

  k_pgen<<<64, 256, 0, stream>>>(ctx, h_prev, embedded, pw, pb, o_pgen);

  dim3 g4(64, 4);
  k_gru<<<g4, 128, 0, stream>>>(embedded, ctx, h_prev, wih, whh, bih, bhh, o_hidden, xb);

  k_outproj<<<(3125 + 7) / 8, 256, 0, stream>>>(xb, outW, outb, logits);
  k_softmax_v<<<64, 256, 0, stream>>>(logits, o_pgen, o_final);
  k_scatter<<<64, 512, 0, stream>>>(ext, o_w, o_pgen, o_final);
  long nf = (long)64 * 50050;
  k_log<<<(int)((nf + 255) / 256), 256, 0, stream>>>(o_final, nf);
}
